// PooledSelfAttention2d_16071767622057
// MI455X (gfx1250) — matrix operand order repaired and checked by reading, not compiled, not run on hardware
//
#include <hip/hip_runtime.h>

typedef __bf16 bf16_t;
typedef __attribute__((ext_vector_type(16))) __bf16          v16bf;
typedef __attribute__((ext_vector_type(8)))  float           v8f;
typedef __attribute__((ext_vector_type(16))) unsigned short  v16u;
typedef unsigned short u16;

// ---------- helpers ----------

__device__ __forceinline__ unsigned short f2bf_rne(float f) {
    unsigned u = __float_as_uint(f);
    return (unsigned short)((u + 0x7FFFu + ((u >> 16) & 1u)) >> 16);
}

// Build a v16bf fragment from two 16-byte chunks (A-fragment style).
__device__ __forceinline__ v16bf frag2(const u16* p0, const u16* p1) {
    struct T { uint4 a, b; } t;
    t.a = *(const uint4*)p0;
    t.b = *(const uint4*)p1;
    return __builtin_bit_cast(v16bf, t);
}
// 16 contiguous bf16 (B-fragment style).
__device__ __forceinline__ v16bf frag16(const u16* p) { return frag2(p, p + 8); }

__device__ __forceinline__ v8f wmma_bf16(v16bf a, v16bf b, v8f c) {
    return __builtin_amdgcn_wmma_f32_16x16x32_bf16(
        false, a, false, b, (short)0, c, false, false);
}

// ---------- kernel 0: convert all weights f32 -> bf16 (one time, 327,680 elems) ----------
// layout in wb: wq [64*512] | wk [64*512] | wv [256*512] | wo [512*256]

__global__ void k_cvt_w(const float* __restrict__ wq, const float* __restrict__ wk,
                        const float* __restrict__ wv, const float* __restrict__ wo,
                        u16* __restrict__ wb) {
    int i = blockIdx.x * 256 + threadIdx.x;   // grid covers 327680
    float v;
    if      (i <  32768) v = wq[i];
    else if (i <  65536) v = wk[i - 32768];
    else if (i < 196608) v = wv[i - 65536];
    else                 v = wo[i - 196608];
    wb[i] = f2bf_rne(v);
}

// ---------- kernel 1: X [b,512,4096] f32 -> XT [b,4096,512] bf16 ----------

__global__ void k_transpose_cast(const float* __restrict__ x, u16* __restrict__ xt) {
    __shared__ u16 tile[64][65];
    int b = blockIdx.z, c0 = blockIdx.y * 64, n0 = blockIdx.x * 64;
    int tid = threadIdx.x;
    int tx = tid & 63, ty = tid >> 6;   // 4 rows per pass
    const float* xp = x + ((size_t)b * 512 + c0) * 4096 + n0;
#pragma unroll
    for (int p = 0; p < 16; ++p) {
        int c = ty + p * 4;
        tile[c][tx] = f2bf_rne(xp[(size_t)c * 4096 + tx]);
    }
    __syncthreads();
    u16* xo = xt + ((size_t)b * 4096 + n0) * 512 + c0;
#pragma unroll
    for (int p = 0; p < 16; ++p) {
        int n = ty + p * 4;
        xo[(size_t)n * 512 + tx] = tile[tx][n];
    }
}

// ---------- kernel 2: projections (mode 0: f, 1: g pooled, 2: h pooled) ----------
// f: [b,4096,64]  g: [b,1024,64]  h: [b,256,1024]   (all bf16)
// W is pre-converted bf16, row-major [Co,512]

__global__ void k_proj(const u16* __restrict__ xt, const u16* __restrict__ W,
                       u16* __restrict__ out, int mode) {
    int lane = threadIdx.x & 31;
    int wv   = threadIdx.x >> 5;
    int wid  = blockIdx.x * 8 + wv;
    int row  = lane & 15, half = lane >> 4;

    if (mode == 0) {
        int b = wid >> 10, rem = wid & 1023;
        int coT = rem >> 8, nT = rem & 255;
        int n0 = nT * 16;
        const u16* wa = W + (size_t)(coT * 16 + row) * 512 + half * 16;
        const u16* xb = xt + ((size_t)b * 4096 + n0 + row) * 512 + half * 16;
        v8f acc = {};
        for (int k = 0; k < 16; ++k) {
            v16bf aw = frag16(wa + k * 32);
            acc = wmma_bf16(aw, frag16(xb + k * 32), acc);
        }
        u16* fp = out + ((size_t)b * 4096 + n0 + row) * 64 + coT * 16 + half * 8;
#pragma unroll
        for (int r = 0; r < 8; r += 2) {
            unsigned pk = (unsigned)f2bf_rne(acc[r]) | ((unsigned)f2bf_rne(acc[r + 1]) << 16);
            *(unsigned*)(fp + r) = pk;
        }
    } else {
        int b, coT, pt;
        if (mode == 1) { b = wid >> 9;  int r = wid & 511;  coT = r >> 7; pt = r & 127; }
        else           { b = wid >> 11; int r = wid & 2047; coT = r >> 7; pt = r & 127; }
        int i = pt >> 2, j = pt & 3;
        int n0a = i * 128 + j * 16;                 // spatial row 2i
        const u16* wa  = W + (size_t)(coT * 16 + row) * 512 + half * 16;
        const u16* xba = xt + ((size_t)b * 4096 + n0a + row) * 512 + half * 16;
        const u16* xbb = xba + (size_t)64 * 512;    // spatial row 2i+1
        v8f da = {}, db = {};
        for (int k = 0; k < 16; ++k) {
            v16bf aw = frag16(wa + k * 32);
            da = wmma_bf16(aw, frag16(xba + k * 32), da);
            db = wmma_bf16(aw, frag16(xbb + k * 32), db);
        }
        float vm[8];
#pragma unroll
        for (int r = 0; r < 8; ++r) {
            float v = fmaxf(da[r], db[r]);          // vertical pool
            vm[r] = fmaxf(v, __shfl_xor(v, 1, 32)); // horizontal pool
        }
        if ((lane & 1) == 0) {
            int m = i * 32 + j * 8 + (row >> 1);
            if (mode == 1) {
                u16* gp = out + ((size_t)b * 1024 + m) * 64 + coT * 16 + half * 8;
#pragma unroll
                for (int r = 0; r < 8; r += 2) {
                    unsigned pk = (unsigned)f2bf_rne(vm[r]) | ((unsigned)f2bf_rne(vm[r + 1]) << 16);
                    *(unsigned*)(gp + r) = pk;
                }
            } else {
#pragma unroll
                for (int r = 0; r < 8; ++r) {
                    int c = coT * 16 + r + 8 * half;
                    out[((size_t)b * 256 + c) * 1024 + m] = f2bf_rne(vm[r]);
                }
            }
        }
    }
}

// ---------- kernel 3: attention, two-pass softmax, o = h * beta^T ----------
// f [b,4096,64], g [b,1024,64], h [b,256,1024] -> o [b,4096,256] (bf16)

__global__ void k_attn(const u16* __restrict__ f, const u16* __restrict__ g,
                       const u16* __restrict__ h, u16* __restrict__ o) {
    __shared__ __align__(16) u16 pbuf_all[8][16][32];
    int lane = threadIdx.x & 31, wv = threadIdx.x >> 5;
    int wid = blockIdx.x * 8 + wv;          // 4096 query tiles total
    int b = wid >> 8, qT = wid & 255;
    int n0 = qT * 16;
    int row = lane & 15, half = lane >> 4;
    u16(*pbuf)[32] = pbuf_all[wv];

    const u16* fp = f + ((size_t)b * 4096 + n0 + row) * 64;
    v16bf af0 = frag16(fp + half * 16);       // c 0..31
    v16bf af1 = frag16(fp + 32 + half * 16);  // c 32..63
    const u16* gb = g + (size_t)b * 1024 * 64;
    const u16* hb = h + (size_t)b * 256 * 1024;

    float rm[8], rs[8];
#pragma unroll
    for (int r = 0; r < 8; ++r) { rm[r] = -1e30f; rs[r] = 0.f; }

    // ---- pass 1: row max / row sum ----
    for (int mc = 0; mc < 32; ++mc) {
        const u16* g0 = gb + (size_t)(mc * 32 + row) * 64 + half * 16;
        v8f d0 = {}, d1 = {};
        d0 = wmma_bf16(af0, frag16(g0), d0);
        d0 = wmma_bf16(af1, frag16(g0 + 32), d0);
        d1 = wmma_bf16(af0, frag16(g0 + 16 * 64), d1);
        d1 = wmma_bf16(af1, frag16(g0 + 16 * 64 + 32), d1);
#pragma unroll
        for (int r = 0; r < 8; ++r) {
            float mx = fmaxf(d0[r], d1[r]);
            if (mx > rm[r]) { rs[r] *= __expf(rm[r] - mx); rm[r] = mx; }
            rs[r] += __expf(d0[r] - rm[r]) + __expf(d1[r] - rm[r]);
        }
    }
    // merge (m,s) across the 16 key-lanes
#pragma unroll
    for (int st = 1; st < 16; st <<= 1) {
#pragma unroll
        for (int r = 0; r < 8; ++r) {
            float om = __shfl_xor(rm[r], st, 32);
            float os = __shfl_xor(rs[r], st, 32);
            float nm = fmaxf(rm[r], om);
            rs[r] = rs[r] * __expf(rm[r] - nm) + os * __expf(om - nm);
            rm[r] = nm;
        }
    }
    float rinv[8];
#pragma unroll
    for (int r = 0; r < 8; ++r) rinv[r] = 1.0f / rs[r];

    // ---- pass 2: normalized probs -> PV accumulate ----
    v8f acc[16];
#pragma unroll
    for (int ct = 0; ct < 16; ++ct) acc[ct] = v8f{};

    for (int mc = 0; mc < 32; ++mc) {
        const u16* g0 = gb + (size_t)(mc * 32 + row) * 64 + half * 16;
        v8f d0 = {}, d1 = {};
        d0 = wmma_bf16(af0, frag16(g0), d0);
        d0 = wmma_bf16(af1, frag16(g0 + 32), d0);
        d1 = wmma_bf16(af0, frag16(g0 + 16 * 64), d1);
        d1 = wmma_bf16(af1, frag16(g0 + 16 * 64 + 32), d1);
#pragma unroll
        for (int r = 0; r < 8; ++r) {
            float p0 = __expf(d0[r] - rm[r]) * rinv[r];
            float p1 = __expf(d1[r] - rm[r]) * rinv[r];
            int M = r + 8 * half;               // query row of this D element
            pbuf[M][row]      = f2bf_rne(p0);   // m = mc*32 + row
            pbuf[M][16 + row] = f2bf_rne(p1);   // m = mc*32 + 16 + row
        }
        asm volatile("s_wait_dscnt 0x0" ::: "memory");  // same-wave LDS RAW
        v16bf bp = frag2(&pbuf[row][half * 16], &pbuf[row][half * 16 + 8]);
#pragma unroll
        for (int ct = 0; ct < 16; ++ct) {
            const u16* hp = hb + (size_t)(ct * 16 + row) * 1024 + mc * 32 + half * 16;
            acc[ct] = wmma_bf16(frag16(hp), bp, acc[ct]);
        }
    }

    u16* op = o + ((size_t)b * 4096 + n0 + row) * 256;
#pragma unroll
    for (int ct = 0; ct < 16; ++ct) {
        int cb = ct * 16 + half * 8;
#pragma unroll
        for (int r = 0; r < 8; r += 2) {
            unsigned pk = (unsigned)f2bf_rne(acc[ct][r]) |
                          ((unsigned)f2bf_rne(acc[ct][r + 1]) << 16);
            *(unsigned*)(op + cb + r) = pk;
        }
    }
}

// ---------- kernel 4: out = gamma * (wo @ o) + x ----------
// wo pre-converted bf16 [512,256]

__global__ void k_out(const u16* __restrict__ o, const u16* __restrict__ wo,
                      const float* __restrict__ x, const float* __restrict__ gamma,
                      float* __restrict__ out) {
    int lane = threadIdx.x & 31, wv = threadIdx.x >> 5;
    int wid = blockIdx.x * 8 + wv;          // 32768 waves
    int row = lane & 15, half = lane >> 4;
    int b = wid >> 11, rem = wid & 2047;
    int coT = rem >> 6, nQ = rem & 63;
    int co0 = coT * 16, n0 = nQ * 64;

    const u16* wp0 = wo + (size_t)(co0 + row) * 256 + half * 16;
    const u16* ob  = o + ((size_t)b * 4096 + n0 + row) * 256 + half * 16;
    v8f acc[4];
#pragma unroll
    for (int q = 0; q < 4; ++q) acc[q] = v8f{};
    for (int k = 0; k < 8; ++k) {
        v16bf aw = frag16(wp0 + k * 32);
#pragma unroll
        for (int q = 0; q < 4; ++q)
            acc[q] = wmma_bf16(aw, frag16(ob + (size_t)q * 16 * 256 + k * 32), acc[q]);
    }
    float gm = gamma[0];
#pragma unroll
    for (int q = 0; q < 4; ++q) {
        size_t base = ((size_t)b * 512 + co0 + half * 8) * 4096 + n0 + q * 16 + row;
        const float* xp = x + base;
        float*       op = out + base;
#pragma unroll
        for (int r = 0; r < 8; ++r)
            op[(size_t)r * 4096] = gm * acc[q][r] + xp[(size_t)r * 4096];
    }
}

// ---------- launcher ----------

extern "C" void kernel_launch(void* const* d_in, const int* in_sizes, int n_in,
                              void* d_out, int out_size, void* d_ws, size_t ws_size,
                              hipStream_t stream) {
    const float* x     = (const float*)d_in[0];
    const float* wq    = (const float*)d_in[1];
    const float* wk    = (const float*)d_in[2];
    const float* wv    = (const float*)d_in[3];
    const float* wo    = (const float*)d_in[4];
    const float* gamma = (const float*)d_in[5];

    char* ws = (char*)d_ws;
    u16* XT = (u16*)ws;                               // 64 MB  [b,4096,512]
    u16* F  = (u16*)(ws + ((size_t)64 << 20));        //  8 MB  [b,4096,64]
    u16* G  = (u16*)(ws + ((size_t)72 << 20));        //  2 MB  [b,1024,64]
    u16* H  = (u16*)(ws + ((size_t)74 << 20));        //  8 MB  [b,256,1024]
    u16* WB = (u16*)(ws + ((size_t)82 << 20));        //  0.65 MB bf16 weights
    u16* O  = XT;                                     // 32 MB, reuses XT region

    u16* WQb = WB;               // 64*512
    u16* WKb = WB + 32768;       // 64*512
    u16* WVb = WB + 65536;       // 256*512
    u16* WOb = WB + 196608;      // 512*256

    k_cvt_w<<<1280, 256, 0, stream>>>(wq, wk, wv, wo, WB);
    k_transpose_cast<<<dim3(64, 8, 16), 256, 0, stream>>>(x, XT);
    k_proj<<<2048, 256, 0, stream>>>(XT, WQb, F, 0);  // 16384 waves
    k_proj<<<1024, 256, 0, stream>>>(XT, WKb, G, 1);  //  8192 waves
    k_proj<<<4096, 256, 0, stream>>>(XT, WVb, H, 2);  // 32768 waves
    k_attn<<<512,  256, 0, stream>>>(F, G, H, O);     //  4096 waves
    k_out <<<4096, 256, 0, stream>>>(O, WOb, x, gamma, (float*)d_out);
}
